// PandGNN_39711267618947
// MI455X (gfx1250) — matrix operand order repaired and checked by reading, not compiled, not run on hardware
//
#include <hip/hip_runtime.h>
#include <hip/hip_bf16.h>
#include <math.h>

typedef __attribute__((ext_vector_type(16))) _Float16 v16h;
typedef __attribute__((ext_vector_type(8)))  float    v8f;

#define MM      100000
#define NITEM   50000
#define NNODES  150000
#define DIM     64
#define NEDGE   2000000
#define BB      8192
#define KK      40
#define REGC    1e-4f

// x0 = concat(E_pos, E_item) row pointer
__device__ __forceinline__ const float* x0_row(const float* Ep, const float* Ei, int node) {
    return (node < MM) ? (Ep + (size_t)node * DIM)
                       : (Ei + (size_t)(node - MM) * DIM);
}

// ---------------- init: zero deg buffer and output scalar -------------------
__global__ void k_init(float* deg, float* out) {
    int i = blockIdx.x * blockDim.x + threadIdx.x;
    if (i < NNODES) deg[i] = 0.f;
    if (i == 0) out[0] = 0.f;
}

// ---------------- degree over col (float counts are exact here) -------------
__global__ void k_deg(const int* __restrict__ col, float* __restrict__ deg) {
    int e = blockIdx.x * blockDim.x + threadIdx.x;
    if (e < NEDGE) atomicAdd(&deg[col[e]], 1.0f);
}

// ---------------- deg -> deg^-1/2 in place ----------------------------------
__global__ void k_dis(float* __restrict__ deg) {
    int i = blockIdx.x * blockDim.x + threadIdx.x;
    if (i < NNODES) {
        float d = deg[i];
        deg[i] = (d > 0.f) ? rsqrtf(d) : 0.f;
    }
}

// ---------------- self term init: xout = dis^2 * x0 (from embeddings) -------
__global__ void k_self0(const float* __restrict__ Ep, const float* __restrict__ Ei,
                        const float* __restrict__ dis, float* __restrict__ xout) {
    int i = blockIdx.x * blockDim.x + threadIdx.x;     // one float4 per thread
    if (i >= NNODES * (DIM / 4)) return;
    int node = i >> 4, q = i & 15;
    float s = dis[node]; s *= s;
    float4 v = ((const float4*)x0_row(Ep, Ei, node))[q];
    float4 r = { s * v.x, s * v.y, s * v.z, s * v.w };
    ((float4*)(xout + (size_t)node * DIM))[q] = r;
}

// ---------------- self term init: xout = dis^2 * xin (from buffer) ----------
__global__ void k_self(const float* __restrict__ xin, const float* __restrict__ dis,
                       float* __restrict__ xout) {
    int i = blockIdx.x * blockDim.x + threadIdx.x;
    if (i >= NNODES * (DIM / 4)) return;
    int node = i >> 4, q = i & 15;
    float s = dis[node]; s *= s;
    float4 v = ((const float4*)(xin + (size_t)node * DIM))[q];
    float4 r = { s * v.x, s * v.y, s * v.z, s * v.w };
    ((float4*)(xout + (size_t)node * DIM))[q] = r;
}

// ---------------- edge scatter: xout[col] += dis[row]*dis[col]*x0[row] ------
__global__ void __launch_bounds__(256)
k_scatter0(const int* __restrict__ rowi, const int* __restrict__ coli,
           const float* __restrict__ dis, const float* __restrict__ Ep,
           const float* __restrict__ Ei, float* __restrict__ xout) {
    int tid  = blockIdx.x * blockDim.x + threadIdx.x;
    int e    = tid >> 5;                 // one wave32 per edge
    int lane = threadIdx.x & 31;
    if (e >= NEDGE) return;
    int r = rowi[e], c = coli[e];
    float nrm = dis[r] * dis[c];
    float2 v = ((const float2*)x0_row(Ep, Ei, r))[lane];   // 2 floats/lane = 64
    float* xo = xout + (size_t)c * DIM + lane * 2;
    atomicAdd(xo,     nrm * v.x);
    atomicAdd(xo + 1, nrm * v.y);
}

__global__ void __launch_bounds__(256)
k_scatter(const int* __restrict__ rowi, const int* __restrict__ coli,
          const float* __restrict__ dis, const float* __restrict__ xin,
          float* __restrict__ xout) {
    int tid  = blockIdx.x * blockDim.x + threadIdx.x;
    int e    = tid >> 5;
    int lane = threadIdx.x & 31;
    if (e >= NEDGE) return;
    int r = rowi[e], c = coli[e];
    float nrm = dis[r] * dis[c];
    float2 v = ((const float2*)(xin + (size_t)r * DIM))[lane];
    float* xo = xout + (size_t)c * DIM + lane * 2;
    atomicAdd(xo,     nrm * v.x);
    atomicAdd(xo + 1, nrm * v.y);
}

// ---------------- z = (x0 + x1 + x2)/3, written into x1 in place ------------
__global__ void k_zavg(const float* __restrict__ Ep, const float* __restrict__ Ei,
                       const float* __restrict__ x2, float* __restrict__ x1) {
    int i = blockIdx.x * blockDim.x + threadIdx.x;
    if (i >= NNODES * (DIM / 4)) return;
    int node = i >> 4, q = i & 15;
    float4 v0 = ((const float4*)x0_row(Ep, Ei, node))[q];
    float4* p1 = (float4*)(x1 + (size_t)node * DIM);
    float4 v1 = p1[q];
    float4 v2 = ((const float4*)(x2 + (size_t)node * DIM))[q];
    const float k = 1.f / 3.f;
    float4 r = { (v0.x + v1.x + v2.x) * k, (v0.y + v1.y + v2.y) * k,
                 (v0.z + v1.z + v2.z) * k, (v0.w + v1.w + v2.w) * k };
    p1[q] = r;
}

__device__ __forceinline__ float log_sigmoid(float x) {
    return fminf(x, 0.f) - log1pf(expf(-fabsf(x)));
}

// ---------------- loss: wave-per-sample; 41 dots via v_wmma_f32_16x16x32_f16
// A tile (16 rows = [v1; n1(40); zero-pad], K=64 as 2 chunks of 32, f16)
// B = u1 replicated in all 16 columns -> C column 0 holds the 16 dots (f32 acc)
__global__ void __launch_bounds__(256)
k_loss(const float* __restrict__ z, const int* __restrict__ u,
       const int* __restrict__ v, const float* __restrict__ w,
       const int* __restrict__ n, float* __restrict__ out) {
    __shared__ float sdots[8 * 48];
    int lane = threadIdx.x & 31;
    int wv   = threadIdx.x >> 5;
    int b    = blockIdx.x * 8 + wv;
    int l16  = lane & 15;
    int hi   = lane >> 4;                       // 0: lanes 0-15, 1: lanes 16-31

    const float* up = z + (size_t)u[b] * DIM;

    // B operand (u1) matching A: lanes 0-15 hold K 0-7 / 16-23, lanes
    // 16-31 hold K 8-15 / 24-31, 2 packed halves per VGPR. Hoisted out of tile loop.
    v16h B0, B1;
    float regu = 0.f;
    {
        int base0 = hi * 8;                     // K-chunk 0 (d 0..31)
        int base1 = 32 + hi * 8;                // K-chunk 1 (d 32..63)
        #pragma unroll
        for (int i = 0; i < 16; i++) {
            float f0 = up[base0 + i + (i >> 3) * 8];
            float f1 = up[base1 + i + (i >> 3) * 8];
            B0[i] = (_Float16)f0;
            B1[i] = (_Float16)f1;
            if (l16 == 0) regu += f0 * f0 + f1 * f1;   // lanes 0,16 cover u1 once
        }
    }

    float regrow = 0.f;
    #pragma unroll
    for (int t3 = 0; t3 < 3; t3++) {
        int t = t3 * 16 + l16;                  // global row id 0..47
        const float* rp = nullptr;
        if (t == 0)        rp = z + (size_t)v[b] * DIM;
        else if (t <= KK)  rp = z + (size_t)n[b * KK + (t - 1)] * DIM;

        v8f c = { 0.f, 0.f, 0.f, 0.f, 0.f, 0.f, 0.f, 0.f };
        #pragma unroll
        for (int chunk = 0; chunk < 2; chunk++) {
            // A layout (16-bit A 16x32): lanes 0-15 hold K 0-7 / 16-23,
            // lanes 16-31 hold K 8-15 / 24-31 (packed 2 per VGPR).
            int abase = chunk * 32 + hi * 8;
            v16h a;
            if (rp) {
                #pragma unroll
                for (int i = 0; i < 8; i++) {
                    float f0 = rp[abase + i];
                    float f1 = rp[abase + 16 + i];
                    a[i]     = (_Float16)f0;
                    a[8 + i] = (_Float16)f1;
                    regrow  += f0 * f0 + f1 * f1;  // lane pair covers row once
                }
            } else {
                #pragma unroll
                for (int i = 0; i < 16; i++) a[i] = (_Float16)0.f;
            }
            v16h bop = chunk ? B1 : B0;
            c = __builtin_amdgcn_wmma_f32_16x16x32_f16(
                    false, a, false, bop, (short)0, c, false, false);
        }
        // C layout: VGPR r -> M=r (lanes 0-15) / M=8+r (lanes 16-31); all
        // columns identical (B replicated), so lanes 0 and 16 extract.
        if (l16 == 0) {
            #pragma unroll
            for (int r = 0; r < 8; r++)
                sdots[wv * 48 + t3 * 16 + hi * 8 + r] = c[r];
        }
    }
    __syncthreads();

    const float* dots = sdots + wv * 48;
    float pos  = dots[0];
    float wb   = w[b];
    float coef = (wb > 0.f) ? 1.f : ((wb < 0.f) ? 3.f : 2.f);
    float cp   = coef * pos;

    float acc = log_sigmoid(cp - dots[1 + lane]);          // k = lane, 0..31
    if (lane < 8) acc += log_sigmoid(cp - dots[33 + lane]); // k = 32..39

    float contrib = REGC * (regrow + ((l16 == 0) ? regu : 0.f)) - acc;
    #pragma unroll
    for (int off = 16; off > 0; off >>= 1)
        contrib += __shfl_xor(contrib, off, 32);
    if (lane == 0) atomicAdd(out, contrib);
}

extern "C" void kernel_launch(void* const* d_in, const int* in_sizes, int n_in,
                              void* d_out, int out_size, void* d_ws, size_t ws_size,
                              hipStream_t stream) {
    (void)in_sizes; (void)n_in; (void)out_size; (void)ws_size;

    const int*   u    = (const int*)d_in[0];
    const int*   v    = (const int*)d_in[1];
    const float* w    = (const float*)d_in[2];
    const int*   n    = (const int*)d_in[3];
    const int*   eidx = (const int*)d_in[4];
    const float* Ep   = (const float*)d_in[5];
    const float* Ei   = (const float*)d_in[6];
    // d_in[7] = EPOCH (==0 -> simple signed-BPR path, unused)

    const int* erow = eidx;           // edge_index[0]
    const int* ecol = eidx + NEDGE;   // edge_index[1]

    // workspace: dis[NN] | x1[NN*64] | x2[NN*64]  (~77.4 MB of floats)
    float* ws  = (float*)d_ws;
    float* dis = ws;
    float* x1  = ws + NNODES;
    float* x2  = x1 + (size_t)NNODES * DIM;
    float* out = (float*)d_out;

    const int TB = 256;
    int nodeBlk = (NNODES + TB - 1) / TB;
    int elemBlk = (NNODES * (DIM / 4) + TB - 1) / TB;
    int edgeThr = (NEDGE + TB - 1) / TB;      // one thread per edge
    int edgeWav = NEDGE / (TB / 32);          // one wave per edge (2M % 8 == 0)

    k_init    <<<nodeBlk, TB, 0, stream>>>(dis, out);
    k_deg     <<<edgeThr, TB, 0, stream>>>(ecol, dis);
    k_dis     <<<nodeBlk, TB, 0, stream>>>(dis);
    // layer 1: x1 = dis^2 * x0 + scatter(norm * x0[row] -> x1[col])
    k_self0   <<<elemBlk, TB, 0, stream>>>(Ep, Ei, dis, x1);
    k_scatter0<<<edgeWav, TB, 0, stream>>>(erow, ecol, dis, Ep, Ei, x1);
    // layer 2: x2 = dis^2 * x1 + scatter(norm * x1[row] -> x2[col])
    k_self    <<<elemBlk, TB, 0, stream>>>(x1, dis, x2);
    k_scatter <<<edgeWav, TB, 0, stream>>>(erow, ecol, dis, x1, x2);
    // z = (x0 + x1 + x2)/3 written into x1
    k_zavg    <<<elemBlk, TB, 0, stream>>>(Ep, Ei, x2, x1);
    // fused gather + WMMA dots + log-sigmoid + reg, reduced into out[0]
    k_loss    <<<BB / 8, TB, 0, stream>>>(x1, u, v, w, n, out);
}
